// ExtraMSABlock_22814866276378
// MI455X (gfx1250) — matrix operand order repaired and checked by reading, not compiled, not run on hardware
//
#include <hip/hip_runtime.h>

// ======================= problem dimensions =======================
#define S_   256
#define R_   192
#define CE_  64
#define CZ_  128
#define HR_  8
#define CR_  8
#define HC_  8
#define CC_  8
#define CO_  32
#define HT_  4
#define CT_  32
#define NT_  4

typedef __attribute__((ext_vector_type(16))) _Float16 v16h;
typedef __attribute__((ext_vector_type(8)))  float    v8f;
typedef __attribute__((ext_vector_type(4)))  float    f4;
typedef __attribute__((ext_vector_type(4)))  _Float16 h4;

// ---- WMMA fragment loaders (ISA 7.12.2 layouts) ------------------
// A (16x32 f16): lane = M%16, K split by lane-half; vgpr i<4 -> K=2i+8*half,
// i>=4 -> K=16+2(i-4)+8*half.  B (32x16): lane = N%16, K in same order as A.
__device__ __forceinline__ v16h afrag_row(const _Float16* p, int half) {
  v16h a;
#pragma unroll
  for (int i = 0; i < 4; i++) {
    int kb = 2 * i + 8 * half;
    a[2 * i] = p[kb]; a[2 * i + 1] = p[kb + 1];
  }
#pragma unroll
  for (int i = 0; i < 4; i++) {
    int kb = 16 + 2 * i + 8 * half;
    a[8 + 2 * i] = p[kb]; a[9 + 2 * i] = p[kb + 1];
  }
  return a;
}
__device__ __forceinline__ v16h bfrag16(const _Float16* p, int ld, int half) {
  v16h b;
#pragma unroll
  for (int i = 0; i < 16; i++) b[i] = p[(i < 8 ? 8 * half + i : 16 + 8 * half + (i - 8)) * ld];
  return b;
}
__device__ __forceinline__ float sigm(float x) { return 1.0f / (1.0f + __expf(-x)); }

// ======================= generic WMMA GEMM ========================
// C[m,n] (+)= act( alpha * sum_k A[m,k]*B[k,n] + bias[n] )
// Requirements (asserted by construction of every call site):
//   K % 32 == 0;  N % 4 == 0;  vector fast path taken iff cs==1.
// act: 0 none, 1 relu, 2 sigmoid, 3 C = sigmoid(x)*C (gating)
// cf16: store C as _Float16 (same element strides)

struct GemmArgs {
  const float* A; const float* B; float* C; const float* bias;
  long long rsA, csA, bsA;
  long long rsB, csB, bsB;
  long long rsC, csC, bsC;
  int M, N, K;
  float alpha;
  int act, accum, cf16;
};

#define BM 128
#define BN 128
#define BK 32

__global__ __launch_bounds__(256) void gemm_wmma(GemmArgs g) {
  __shared__ _Float16 As[2][BM][BK];   // 2 x 8 KB
  __shared__ _Float16 Bs[2][BK][BN];   // 2 x 8 KB

  const float* A = g.A + (long long)blockIdx.z * g.bsA;
  const float* B = g.B + (long long)blockIdx.z * g.bsB;
  float*       C = g.C + (long long)blockIdx.z * g.bsC;

  const int m0 = blockIdx.x * BM;
  const int n0 = blockIdx.y * BN;
  const int tid  = threadIdx.x;
  const int lane = tid & 31, wid = tid >> 5;
  const int wm = (wid & 3) * 32;       // 4x2 wave grid, wave tile 32x64
  const int wn = (wid >> 2) * 64;
  const int half = lane >> 4, l16 = lane & 15;

  v8f acc[2][4] = {};
  const int nk = g.K >> 5;             // K % 32 == 0

  auto stageA = [&](int kt, int buf) {
    const int k0 = kt * BK;
    if (g.csA == 1) {                  // contiguous-K fast path, 128-bit loads
#pragma unroll
      for (int it = 0; it < 4; it++) {
        int vi = tid + 256 * it;       // 1024 float4 loads
        int m = vi >> 3, kq = (vi & 7) << 2;
        int gm = m0 + m;
        bool ok = gm < g.M;
        const f4 zero = {0.f, 0.f, 0.f, 0.f};
        f4 val = *(const f4*)(A + (long long)(ok ? gm : g.M - 1) * g.rsA + (k0 + kq));
        if (!ok) val = zero;
        h4 hv = {(_Float16)val[0], (_Float16)val[1], (_Float16)val[2], (_Float16)val[3]};
        *(h4*)&As[buf][m][kq] = hv;
      }
    } else {                           // strided (transposed/batched) fallback
      for (int idx = tid; idx < BM * BK; idx += 256) {
        int m = idx >> 5, k = idx & 31;
        int gm = m0 + m;
        bool ok = gm < g.M;
        float val = A[(long long)(ok ? gm : g.M - 1) * g.rsA + (long long)(k0 + k) * g.csA];
        As[buf][m][k] = (_Float16)(ok ? val : 0.f);
      }
    }
  };
  auto stageB = [&](int kt, int buf) {
    const int k0 = kt * BK;
    if (g.csB == 1) {
#pragma unroll
      for (int it = 0; it < 4; it++) {
        int vi = tid + 256 * it;       // 1024 float4 loads
        int k = vi >> 5, nq = (vi & 31) << 2;
        int gn = n0 + nq;
        bool ok = gn < g.N;            // N % 4 == 0 -> whole quad valid/invalid
        const f4 zero = {0.f, 0.f, 0.f, 0.f};
        f4 val = *(const f4*)(B + (long long)(k0 + k) * g.rsB + (ok ? gn : 0));
        if (!ok) val = zero;
        h4 hv = {(_Float16)val[0], (_Float16)val[1], (_Float16)val[2], (_Float16)val[3]};
        *(h4*)&Bs[buf][k][nq] = hv;
      }
    } else {
      for (int idx = tid; idx < BK * BN; idx += 256) {
        int k = idx >> 7, n = idx & 127;
        int gn = n0 + n;
        bool ok = gn < g.N;
        float val = B[(long long)(k0 + k) * g.rsB + (long long)(ok ? gn : 0) * g.csB];
        Bs[buf][k][n] = (_Float16)(ok ? val : 0.f);
      }
    }
  };

  stageA(0, 0); stageB(0, 0);
  __syncthreads();
  for (int kt = 0; kt < nk; kt++) {
    const int cur = kt & 1;
    if (kt + 1 < nk) { stageA(kt + 1, cur ^ 1); stageB(kt + 1, cur ^ 1); }
    v16h a0 = afrag_row(&As[cur][wm + l16][0], half);
    v16h a1 = afrag_row(&As[cur][wm + 16 + l16][0], half);
#pragma unroll
    for (int sn = 0; sn < 4; sn++) {
      v16h b = bfrag16(&Bs[cur][0][wn + sn * 16 + l16], BN, half);
      acc[0][sn] = __builtin_amdgcn_wmma_f32_16x16x32_f16(
          false, a0, false, b, (short)0, acc[0][sn], false, false);
      acc[1][sn] = __builtin_amdgcn_wmma_f32_16x16x32_f16(
          false, a1, false, b, (short)0, acc[1][sn], false, false);
    }
    __syncthreads();
  }

  // epilogue; C tile layout: vgpr r -> M = r + 8*half, N = lane&15
#pragma unroll
  for (int sm = 0; sm < 2; sm++) {
#pragma unroll
    for (int sn = 0; sn < 4; sn++) {
#pragma unroll
      for (int r = 0; r < 8; r++) {
        int row = m0 + wm + sm * 16 + r + 8 * half;
        int col = n0 + wn + sn * 16 + l16;
        if (row < g.M && col < g.N) {
          float v = acc[sm][sn][r] * g.alpha;
          if (g.bias) v += g.bias[col];
          if (g.act == 1) v = fmaxf(v, 0.0f);
          else if (g.act == 2) v = sigm(v);
          long long off = (long long)row * g.rsC + (long long)col * g.csC;
          if (g.cf16)          ((_Float16*)C)[off] = (_Float16)v;
          else if (g.act == 3) C[off] = sigm(v) * C[off];
          else if (g.accum)    C[off] += v;
          else                 C[off] = v;
        }
      }
    }
  }
}

// ======================= LayerNorm (one wave / row) ===============
__global__ __launch_bounds__(256) void ln_kernel(const float* __restrict__ x,
                                                 const float* __restrict__ gam,
                                                 const float* __restrict__ bet,
                                                 float* __restrict__ y,
                                                 int rows, int C) {
  int row = ((int)blockIdx.x * blockDim.x + threadIdx.x) >> 5;
  int lane = threadIdx.x & 31;
  if (row >= rows) return;
  const float* xr = x + (long long)row * C;
  float s = 0.f, s2 = 0.f;
  for (int c = lane; c < C; c += 32) { float v = xr[c]; s += v; s2 += v * v; }
#pragma unroll
  for (int o = 16; o; o >>= 1) { s += __shfl_xor(s, o, 32); s2 += __shfl_xor(s2, o, 32); }
  float mu = s / C;
  float var = s2 / C - mu * mu;
  float inv = rsqrtf(var + 1e-5f);
  float* yr = y + (long long)row * C;
  for (int c = lane; c < C; c += 32) yr[c] = (xr[c] - mu) * inv * gam[c] + bet[c];
}

// ================== fused MSA row attention (per s,h) =============
__global__ __launch_bounds__(256) void row_attn_fused(
    const float* __restrict__ q, const float* __restrict__ k,
    const float* __restrict__ v, const float* __restrict__ bpair,
    float* __restrict__ o) {
  int s = blockIdx.x >> 3;
  int h = blockIdx.x & 7;
  __shared__ float qs[R_][CR_], ks[R_][CR_], vs[R_][CR_];
  int tid = threadIdx.x;
  for (int idx = tid; idx < R_ * CR_; idx += 256) {
    int i = idx >> 3, c = idx & 7;
    long long base = ((long long)s * R_ + i) * (HR_ * CR_) + h * CR_ + c;
    qs[i][c] = q[base]; ks[i][c] = k[base]; vs[i][c] = v[base];
  }
  __syncthreads();
  int lane = tid & 31, w = tid >> 5;
  const float scale = 0.3535533905932738f;   // 1/sqrt(8)
  for (int i = w; i < R_; i += 8) {
    float sc[6]; float mx = -1e30f;
#pragma unroll
    for (int t = 0; t < 6; t++) {
      int j = lane + 32 * t;
      float d = 0.f;
#pragma unroll
      for (int c = 0; c < 8; c++) d += qs[i][c] * ks[j][c];
      d = d * scale + bpair[((long long)i * R_ + j) * HR_ + h];
      sc[t] = d; mx = fmaxf(mx, d);
    }
#pragma unroll
    for (int off = 16; off; off >>= 1) mx = fmaxf(mx, __shfl_xor(mx, off, 32));
    float sum = 0.f;
#pragma unroll
    for (int t = 0; t < 6; t++) { sc[t] = __expf(sc[t] - mx); sum += sc[t]; }
#pragma unroll
    for (int off = 16; off; off >>= 1) sum += __shfl_xor(sum, off, 32);
    float inv = 1.0f / sum;
    float oa[8] = {0, 0, 0, 0, 0, 0, 0, 0};
#pragma unroll
    for (int t = 0; t < 6; t++) {
      int j = lane + 32 * t;
      float p = sc[t];
#pragma unroll
      for (int c = 0; c < 8; c++) oa[c] += p * vs[j][c];
    }
#pragma unroll
    for (int c = 0; c < 8; c++)
#pragma unroll
      for (int off = 16; off; off >>= 1) oa[c] += __shfl_xor(oa[c], off, 32);
    if (lane == 0) {
      long long base = ((long long)s * R_ + i) * (HR_ * CR_) + h * CR_;
#pragma unroll
      for (int c = 0; c < 8; c++) o[base + c] = oa[c] * inv;
    }
  }
}

// ============ fused MSA column global attention (per r) ===========
__global__ __launch_bounds__(256) void col_attn_fused(
    const float* __restrict__ qc, const float* __restrict__ kc,
    const float* __restrict__ vc, float* __restrict__ oc) {
  int r = blockIdx.x;
  __shared__ float qsh[HC_ * CC_];
  __shared__ float ssc[HC_][S_];
  __shared__ float vsh[S_][CC_];
  int tid = threadIdx.x;                 // tid == s index (S_==256)
  if (tid < HC_ * CC_) qsh[tid] = qc[r * (HC_ * CC_) + tid];
  float kv[CC_];
  {
    long long base = ((long long)tid * R_ + r) * CC_;
#pragma unroll
    for (int c = 0; c < CC_; c++) { kv[c] = kc[base + c]; vsh[tid][c] = vc[base + c]; }
  }
  __syncthreads();
#pragma unroll
  for (int hh = 0; hh < HC_; hh++) {
    float d = 0.f;
#pragma unroll
    for (int c = 0; c < CC_; c++) d += qsh[hh * CC_ + c] * kv[c];
    ssc[hh][tid] = d;
  }
  __syncthreads();
  int lane = tid & 31, w = tid >> 5;     // wave w handles head h=w
  float vals[8]; float mx = -1e30f;
#pragma unroll
  for (int t = 0; t < 8; t++) { vals[t] = ssc[w][lane + 32 * t]; mx = fmaxf(mx, vals[t]); }
#pragma unroll
  for (int off = 16; off; off >>= 1) mx = fmaxf(mx, __shfl_xor(mx, off, 32));
  float sum = 0.f;
#pragma unroll
  for (int t = 0; t < 8; t++) { vals[t] = __expf(vals[t] - mx); sum += vals[t]; }
#pragma unroll
  for (int off = 16; off; off >>= 1) sum += __shfl_xor(sum, off, 32);
  float inv = 1.0f / sum;
  float oa[8] = {0, 0, 0, 0, 0, 0, 0, 0};
#pragma unroll
  for (int t = 0; t < 8; t++) {
    int sIdx = lane + 32 * t; float p = vals[t];
#pragma unroll
    for (int c = 0; c < 8; c++) oa[c] += p * vsh[sIdx][c];
  }
#pragma unroll
  for (int c = 0; c < 8; c++)
#pragma unroll
    for (int off = 16; off; off >>= 1) oa[c] += __shfl_xor(oa[c], off, 32);
  if (lane == 0)
#pragma unroll
    for (int c = 0; c < 8; c++) oc[r * (HC_ * CC_) + w * CC_ + c] = oa[c] * inv;
}

// ========== WMMA fused triangle attention (per h,i) ===============
// q,k,v are f16 (written by the projection GEMMs with cf16=1).
// Scores computed by WMMA into f32 LDS (96 rows at a time), softmaxed,
// re-packed as f16 aliased into the score rows, AV again by WMMA.
#define JB 96                           // row-half size
__global__ __launch_bounds__(256) void tri_attn_wmma(
    const _Float16* __restrict__ q, const _Float16* __restrict__ k,
    const _Float16* __restrict__ v, const float* __restrict__ bpair,
    float* __restrict__ o) {
  int i = blockIdx.x >> 2;
  int h = blockIdx.x & 3;
  __shared__ _Float16 qh[R_][CT_];      // 12 KB
  __shared__ _Float16 khT[CT_][R_];     // 12 KB (transposed: B-frag contiguous)
  __shared__ _Float16 vh[R_][CT_];      // 12 KB
  __shared__ float    scs[JB][R_];      // 72 KB (f16 probs aliased in-place)
  int tid = threadIdx.x;
  int lane = tid & 31, wid = tid >> 5;
  int half = lane >> 4, l16 = lane & 15;

  for (int idx = tid; idx < R_ * CT_; idx += 256) {
    int j = idx >> 5, c = idx & 31;
    long long base = ((long long)i * R_ + j) * (HT_ * CT_) + h * CT_ + c;
    qh[j][c] = q[base];
    khT[c][j] = k[base];
    vh[j][c] = v[base];
  }
  __syncthreads();

  const float scale = 0.17677669529663687f;   // 1/sqrt(32)
  for (int jh = 0; jh < 2; jh++) {
    const int j0 = jh * JB;
    // ---- scores: 6 x 12 tiles of 16x16, K = 32 (one WMMA each) ----
    for (int t = wid; t < 72; t += 8) {
      int tj = t / 12, tk = t % 12;
      v16h a = afrag_row(&qh[j0 + tj * 16 + l16][0], half);
      v16h b = bfrag16(&khT[0][tk * 16 + l16], R_, half);
      v8f cacc = {};
      cacc = __builtin_amdgcn_wmma_f32_16x16x32_f16(
          false, a, false, b, (short)0, cacc, false, false);
#pragma unroll
      for (int r = 0; r < 8; r++) {
        int jj = tj * 16 + r + 8 * half;          // local row
        int kk = tk * 16 + l16;
        scs[jj][kk] = cacc[r] * scale +
                      bpair[((long long)(j0 + jj) * R_ + kk) * HT_ + h];
      }
    }
    __syncthreads();
    // ---- softmax rows, re-pack probs as f16 in-place ----
    for (int jj = wid; jj < JB; jj += 8) {
      float sc[6]; float mx = -1e30f;
#pragma unroll
      for (int t = 0; t < 6; t++) { sc[t] = scs[jj][lane + 32 * t]; mx = fmaxf(mx, sc[t]); }
#pragma unroll
      for (int off = 16; off; off >>= 1) mx = fmaxf(mx, __shfl_xor(mx, off, 32));
      float sum = 0.f;
#pragma unroll
      for (int t = 0; t < 6; t++) { sc[t] = __expf(sc[t] - mx); sum += sc[t]; }
#pragma unroll
      for (int off = 16; off; off >>= 1) sum += __shfl_xor(sum, off, 32);
      float inv = 1.0f / sum;
      _Float16* pf = (_Float16*)&scs[jj][0];      // alias: read-before-write
#pragma unroll
      for (int t = 0; t < 6; t++) pf[lane + 32 * t] = (_Float16)(sc[t] * inv);
    }
    __syncthreads();
    // ---- AV: 6 x 2 tiles of 16x16, K = 192 (6 WMMA steps) ----
    for (int t = wid; t < 12; t += 8) {
      int tj = t >> 1, tc = t & 1;
      v8f oacc = {};
      for (int kk = 0; kk < 6; kk++) {
        const _Float16* pa = (const _Float16*)&scs[tj * 16 + l16][0] + kk * 32;
        v16h a = afrag_row(pa, half);
        v16h b = bfrag16(&vh[kk * 32][tc * 16 + l16], CT_, half);
        oacc = __builtin_amdgcn_wmma_f32_16x16x32_f16(
            false, a, false, b, (short)0, oacc, false, false);
      }
#pragma unroll
      for (int r = 0; r < 8; r++) {
        int j = j0 + tj * 16 + r + 8 * half;
        int c = tc * 16 + l16;
        o[((long long)i * R_ + j) * (HT_ * CT_) + h * CT_ + c] = oacc[r];
      }
    }
    __syncthreads();
  }
}

// ======================= elementwise helpers ======================
__global__ void add_kernel(float* dst, const float* src, long long n) {
  long long i = (long long)blockIdx.x * blockDim.x + threadIdx.x;
  long long st = (long long)gridDim.x * blockDim.x;
  for (; i < n; i += st) dst[i] += src[i];
}
__global__ void mul_bcast_kernel(float* g, const float* o, long long n) {
  long long i = (long long)blockIdx.x * blockDim.x + threadIdx.x;
  long long st = (long long)gridDim.x * blockDim.x;
  for (; i < n; i += st) g[i] *= o[i % (long long)(R_ * 64)];
}
__global__ void transpose_rr_kernel(const float* in, float* out) {
  long long n = (long long)R_ * R_ * CZ_;
  long long i = (long long)blockIdx.x * blockDim.x + threadIdx.x;
  long long st = (long long)gridDim.x * blockDim.x;
  for (; i < n; i += st) {
    int c = (int)(i % CZ_);
    long long ij = i / CZ_;
    int j = (int)(ij % R_), ii = (int)(ij / R_);
    out[((long long)j * R_ + ii) * CZ_ + c] = in[i];
  }
}
__global__ void add_transposed_kernel(float* z, const float* u) {
  long long n = (long long)R_ * R_ * CZ_;
  long long i = (long long)blockIdx.x * blockDim.x + threadIdx.x;
  long long st = (long long)gridDim.x * blockDim.x;
  for (; i < n; i += st) {
    int c = (int)(i % CZ_);
    long long ij = i / CZ_;
    int j = (int)(ij % R_), ii = (int)(ij / R_);
    z[i] += u[((long long)j * R_ + ii) * CZ_ + c];
  }
}
__global__ void mean_s_kernel(const float* m, float* mbar) {
  int x = blockIdx.x * blockDim.x + threadIdx.x;
  if (x >= R_ * CE_) return;
  float s = 0.f;
  for (int t = 0; t < S_; t++) s += m[(long long)t * (R_ * CE_) + x];
  mbar[x] = s * (1.0f / S_);
}
__global__ void opm_reorder_kernel(const float* och, float* X, int CI) {
  long long n = (long long)CI * R_ * CO_ * CO_;
  long long i = (long long)blockIdx.x * blockDim.x + threadIdx.x;
  long long st = (long long)gridDim.x * blockDim.x;
  for (; i < n; i += st) {
    int d = (int)(i & 31);
    int c = (int)((i >> 5) & 31);
    long long jc = i >> 10;
    int j = (int)(jc % R_);
    int ic = (int)(jc / R_);
    X[i] = och[((long long)(ic * CO_ + c)) * (R_ * CO_) + j * CO_ + d];
  }
}

// ======================= host-side helpers ========================
static void gemm(hipStream_t st,
                 const float* A, long long rsA, long long csA, long long bsA,
                 const float* B, long long rsB, long long csB, long long bsB,
                 float* C, long long rsC, long long csC, long long bsC,
                 const float* bias, int M, int N, int K, int batch,
                 float alpha, int act, int accum, int cf16 = 0) {
  GemmArgs g{A, B, C, bias, rsA, csA, bsA, rsB, csB, bsB, rsC, csC, bsC,
             M, N, K, alpha, act, accum, cf16};
  dim3 grid((M + BM - 1) / BM, (N + BN - 1) / BN, batch);
  hipLaunchKernelGGL(gemm_wmma, grid, dim3(256), 0, st, g);
}
static void dense(hipStream_t st, const float* A, const float* W, float* C,
                  const float* bias, int M, int N, int K,
                  float alpha, int act, int accum, int cf16 = 0) {
  gemm(st, A, K, 1, 0, W, N, 1, 0, C, N, 1, 0, bias, M, N, K, 1,
       alpha, act, accum, cf16);
}
static void layernorm(hipStream_t st, const float* x, const float* g,
                      const float* b, float* y, int rows, int C) {
  hipLaunchKernelGGL(ln_kernel, dim3((rows + 7) / 8), dim3(256), 0, st,
                     x, g, b, y, rows, C);
}

struct TaP { const float *b_g, *b_o, *ln_g, *ln_b, *w_b, *w_g, *w_k, *w_o, *w_q, *w_v; };
struct TmP { const float *b_ag, *b_ap, *b_bg, *b_bp, *b_g, *b_o, *ln_g, *ln_b,
                         *lno_g, *lno_b, *w_ag, *w_ap, *w_bg, *w_bp, *w_g, *w_o; };

// ======================= top-level launch =========================
// Params arrive as the JAX pytree flatten of setup_inputs(): dict keys in
// sorted order at every level, LN tuples as (gamma, beta).
// Workspace (f32 elements), assumed ws_size >= ~167 MB:
//   big : 18,874,368 ; s1-s5 : 4,718,592 each ; s6 : 1,048,576
extern "C" void kernel_launch(void* const* d_in, const int* in_sizes, int n_in,
                              void* d_out, int out_size, void* d_ws, size_t ws_size,
                              hipStream_t stream) {
  (void)in_sizes; (void)n_in; (void)out_size; (void)ws_size;

  const float* e_in = (const float*)d_in[0];
  const float* z_in = (const float*)d_in[1];
  int idx = 2;
  auto P = [&]() { return (const float*)d_in[idx++]; };

  const float *col_b_g = P(), *col_b_o = P(), *col_ln_g = P(), *col_ln_b = P(),
              *col_w_g = P(), *col_w_k = P(), *col_w_o = P(), *col_w_q = P(),
              *col_w_v = P();
  const float *opm_b_a = P(), *opm_b_b = P(), *opm_b_o = P(), *opm_ln_g = P(),
              *opm_ln_b = P(), *opm_w_a = P(), *opm_w_b = P(), *opm_w_o = P();
  const float *ptr_b1 = P(), *ptr_b2 = P(), *ptr_ln_g = P(), *ptr_ln_b = P(),
              *ptr_w1 = P(), *ptr_w2 = P();
  const float *row_b_g = P(), *row_b_o = P(), *row_ln_g = P(), *row_ln_b = P(),
              *row_lnz_g = P(), *row_lnz_b = P(), *row_w_b = P(), *row_w_g = P(),
              *row_w_k = P(), *row_w_o = P(), *row_w_q = P(), *row_w_v = P();
  TaP taE{P(), P(), P(), P(), P(), P(), P(), P(), P(), P()};
  TaP taS{P(), P(), P(), P(), P(), P(), P(), P(), P(), P()};
  TmP tmI{P(), P(), P(), P(), P(), P(), P(), P(), P(), P(), P(), P(), P(), P(), P(), P()};
  TmP tmO{P(), P(), P(), P(), P(), P(), P(), P(), P(), P(), P(), P(), P(), P(), P(), P()};
  const float *tr_b1 = P(), *tr_b2 = P(), *tr_ln_g = P(), *tr_ln_b = P(),
              *tr_w1 = P(), *tr_w2 = P();

  const long long SZ_E = (long long)S_ * R_ * CE_;
  const long long SZ_Z = (long long)R_ * R_ * CZ_;
  float* e = (float*)d_out;
  float* z = (float*)d_out + SZ_E;
  hipMemcpyAsync(e, e_in, sizeof(float) * SZ_E, hipMemcpyDeviceToDevice, stream);
  hipMemcpyAsync(z, z_in, sizeof(float) * SZ_Z, hipMemcpyDeviceToDevice, stream);

  float* w = (float*)d_ws;
  float* big = w;
  float* s1 = w + 18874368;
  float* s2 = s1 + 4718592;
  float* s3 = s2 + 4718592;
  float* s4 = s3 + 4718592;
  float* s5 = s4 + 4718592;
  float* s6 = s5 + 4718592;

  const int SR = S_ * R_;     // 49152
  const int RR = R_ * R_;     // 36864
  dim3 ew(2048), tb(256);

  // ============ 1) MSA row attention with pair bias ============
  layernorm(stream, e, row_ln_g, row_ln_b, s1, SR, CE_);             // m
  layernorm(stream, z, row_lnz_g, row_lnz_b, big, RR, CZ_);          // ln_z
  dense(stream, big, row_w_b, s6, nullptr, RR, HR_, CZ_, 1.f, 0, 0); // bpair[i,j,h]
  dense(stream, s1, row_w_q, s2, nullptr, SR, 64, 64, 1.f, 0, 0);
  dense(stream, s1, row_w_k, s3, nullptr, SR, 64, 64, 1.f, 0, 0);
  dense(stream, s1, row_w_v, s4, nullptr, SR, 64, 64, 1.f, 0, 0);
  hipLaunchKernelGGL(row_attn_fused, dim3(S_ * HR_), tb, 0, stream, s2, s3, s4, s6, s5);
  dense(stream, s1, row_w_g, s5, row_b_g, SR, 64, 64, 1.f, 3, 0);    // gate o
  dense(stream, s5, row_w_o, e, row_b_o, SR, CE_, 64, 1.f, 0, 1);    // e += proj

  // ============ 2) MSA column global attention ============
  layernorm(stream, e, col_ln_g, col_ln_b, s1, SR, CE_);
  float* mbar = s6;               // R*64
  float* qcol = s6 + 16384;
  float* ocol = s6 + 32768;
  float* kcol = s6 + 49152;       // S*R*8
  float* vcol = kcol + 393216;
  hipLaunchKernelGGL(mean_s_kernel, dim3((R_ * CE_ + 255) / 256), tb, 0, stream, s1, mbar);
  dense(stream, mbar, col_w_q, qcol, nullptr, R_, 64, 64, 0.3535533905932738f, 0, 0);
  dense(stream, s1, col_w_k, kcol, nullptr, SR, CC_, 64, 1.f, 0, 0);
  dense(stream, s1, col_w_v, vcol, nullptr, SR, CC_, 64, 1.f, 0, 0);
  hipLaunchKernelGGL(col_attn_fused, dim3(R_), tb, 0, stream, qcol, kcol, vcol, ocol);
  dense(stream, s1, col_w_g, s2, col_b_g, SR, 64, 64, 1.f, 2, 0);    // g = sigmoid
  hipLaunchKernelGGL(mul_bcast_kernel, ew, tb, 0, stream, s2, ocol, (long long)SR * 64);
  dense(stream, s2, col_w_o, e, col_b_o, SR, CE_, 64, 1.f, 0, 1);

  // ============ 3) MSA transition ============
  layernorm(stream, e, tr_ln_g, tr_ln_b, s1, SR, CE_);
  dense(stream, s1, tr_w1, big, tr_b1, SR, NT_ * CE_, CE_, 1.f, 1, 0);
  dense(stream, big, tr_w2, e, tr_b2, SR, CE_, NT_ * CE_, 1.f, 0, 1);

  // ============ 4) Outer product mean ============
  layernorm(stream, e, opm_ln_g, opm_ln_b, s1, SR, CE_);
  dense(stream, s1, opm_w_a, s2, opm_b_a, SR, CO_, CE_, 1.f, 0, 0);  // a[s,i,c]
  dense(stream, s1, opm_w_b, s3, opm_b_b, SR, CO_, CE_, 1.f, 0, 0);  // b[s,j,d]
  {
    const int CI = 32;
    float* och = big;               // CI*CO x R*CO
    float* X   = big + 6291456;     // CI*R  x CO*CO
    for (int i0 = 0; i0 < R_; i0 += CI) {
      gemm(stream, s2 + (long long)i0 * CO_, 1, (long long)R_ * CO_, 0,
           s3, (long long)R_ * CO_, 1, 0,
           och, (long long)R_ * CO_, 1, 0,
           nullptr, CI * CO_, R_ * CO_, S_, 1, 1.0f / S_, 0, 0);
      hipLaunchKernelGGL(opm_reorder_kernel, dim3(4096), tb, 0, stream, och, X, CI);
      gemm(stream, X, CO_ * CO_, 1, 0, opm_w_o, CZ_, 1, 0,
           z + (long long)i0 * R_ * CZ_, CZ_, 1, 0,
           opm_b_o, CI * R_, CZ_, CO_ * CO_, 1, 1.f, 0, 1);       // z += ...
    }
  }

  // ============ 5/6) Triangle multiplications ============
  for (int pass = 0; pass < 2; pass++) {
    const TmP& tp = pass == 0 ? tmO : tmI;
    bool outgoing = (pass == 0);
    layernorm(stream, z, tp.ln_g, tp.ln_b, s1, RR, CZ_);
    dense(stream, s1, tp.w_ap, s2, tp.b_ap, RR, CZ_, CZ_, 1.f, 0, 0);
    dense(stream, s1, tp.w_ag, s2, tp.b_ag, RR, CZ_, CZ_, 1.f, 3, 0);  // a = sig*p
    dense(stream, s1, tp.w_bp, s3, tp.b_bp, RR, CZ_, CZ_, 1.f, 0, 0);
    dense(stream, s1, tp.w_bg, s3, tp.b_bg, RR, CZ_, CZ_, 1.f, 3, 0);  // b = sig*p
    if (outgoing)  // o[i,j,c] = sum_k a[i,k,c] b[j,k,c]   (128 batched GEMMs)
      gemm(stream, s2, (long long)R_ * CZ_, CZ_, 1,
           s3, CZ_, (long long)R_ * CZ_, 1,
           s4, (long long)R_ * CZ_, CZ_, 1,
           nullptr, R_, R_, R_, CZ_, 1.f, 0, 0);
    else           // o[i,j,c] = sum_k a[k,i,c] b[k,j,c]
      gemm(stream, s2, CZ_, (long long)R_ * CZ_, 1,
           s3, (long long)R_ * CZ_, CZ_, 1,
           s4, (long long)R_ * CZ_, CZ_, 1,
           nullptr, R_, R_, R_, CZ_, 1.f, 0, 0);
    layernorm(stream, s4, tp.lno_g, tp.lno_b, s5, RR, CZ_);
    dense(stream, s5, tp.w_o, s2, tp.b_o, RR, CZ_, CZ_, 1.f, 0, 0);
    dense(stream, s1, tp.w_g, s2, tp.b_g, RR, CZ_, CZ_, 1.f, 3, 0);    // gate
    hipLaunchKernelGGL(add_kernel, ew, tb, 0, stream, z, s2, SZ_Z);
  }

  // ============ 7/8) Triangle attentions (start / end) ============
  for (int pass = 0; pass < 2; pass++) {
    const TaP& tp = pass == 0 ? taS : taE;
    bool trans = (pass == 1);
    const float* zsrc = z;
    if (trans) {
      hipLaunchKernelGGL(transpose_rr_kernel, ew, tb, 0, stream, z, s5);
      zsrc = s5;
    }
    layernorm(stream, zsrc, tp.ln_g, tp.ln_b, s1, RR, CZ_);
    dense(stream, s1, tp.w_b, s6, nullptr, RR, HT_, CZ_, 1.f, 0, 0);   // bpair[j,k,h]
    dense(stream, s1, tp.w_q, s2, nullptr, RR, 128, 128, 1.f, 0, 0, 1); // f16 q
    dense(stream, s1, tp.w_k, s3, nullptr, RR, 128, 128, 1.f, 0, 0, 1); // f16 k
    dense(stream, s1, tp.w_v, s4, nullptr, RR, 128, 128, 1.f, 0, 0, 1); // f16 v
    hipLaunchKernelGGL(tri_attn_wmma, dim3(HT_ * R_), tb, 0, stream,
                       (const _Float16*)s2, (const _Float16*)s3,
                       (const _Float16*)s4, s6, s5);
    dense(stream, s1, tp.w_g, s5, tp.b_g, RR, 128, 128, 1.f, 3, 0);    // gate o
    if (!trans) {
      dense(stream, s5, tp.w_o, z, tp.b_o, RR, CZ_, 128, 1.f, 0, 1);   // z += proj
    } else {
      dense(stream, s5, tp.w_o, big, tp.b_o, RR, CZ_, 128, 1.f, 0, 0);
      hipLaunchKernelGGL(add_transposed_kernel, ew, tb, 0, stream, z, big);
    }
  }

  // ============ 9) Pair transition ============
  layernorm(stream, z, ptr_ln_g, ptr_ln_b, s1, RR, CZ_);
  dense(stream, s1, ptr_w1, big, ptr_b1, RR, NT_ * CZ_, CZ_, 1.f, 1, 0);
  dense(stream, big, ptr_w2, z, ptr_b2, RR, CZ_, NT_ * CZ_, 1.f, 0, 1);
}
